// QLSTM__gen308_65481071407702
// MI455X (gfx1250) — matrix operand order repaired and checked by reading, not compiled, not run on hardware
//
#include <hip/hip_runtime.h>

typedef __attribute__((ext_vector_type(16))) _Float16 v16h;
typedef __attribute__((ext_vector_type(8)))  _Float16 v8h;
typedef __attribute__((ext_vector_type(8)))  float    v8f;

#define T_STEPS 512
#define BATCH   64
#define IN_DIM  512
#define HID     512
#define KDIM    1024          // I + H
#define NWG     32            // H / HSLICE workgroups
#define HSLICE  16            // H columns owned per workgroup
#define THREADS 256           // 8 wave32
#define KP      1032          // padded LDS row stride in f16 elems (1024 + 8)
#define OP      68            // padded LDS row stride for f32 gate tile (64 + 4)

__device__ __forceinline__ float sig_(float x)  { return 1.0f / (1.0f + __expf(-x)); }
__device__ __forceinline__ float tanh_(float x) { return 1.0f - 2.0f / (__expf(2.0f * x) + 1.0f); }

// ---------------------------------------------------------------------------
// Init: zero both h double-buffers (f16) and the grid-barrier counter.
// ---------------------------------------------------------------------------
__global__ void lstm_init_kernel(_Float16* __restrict__ hbuf, int* __restrict__ cnt) {
    int gid = blockIdx.x * blockDim.x + threadIdx.x;
    if (gid < 2 * BATCH * HID) hbuf[gid] = (_Float16)0.0f;
    if (gid == 0) *cnt = 0;
}

// ---------------------------------------------------------------------------
// Persistent LSTM kernel. Workgroup wg owns H columns [wg*16, wg*16+16) for
// all 4 gates (N = 64). Weights live in LDS as f16 for the whole launch.
// ---------------------------------------------------------------------------
__global__ void __launch_bounds__(THREADS, 1)
lstm_persistent_kernel(const float* __restrict__ x,
                       const float* __restrict__ Wf, const float* __restrict__ bf,
                       const float* __restrict__ Wi, const float* __restrict__ bi,
                       const float* __restrict__ Wg, const float* __restrict__ bg,
                       const float* __restrict__ Wo, const float* __restrict__ bo,
                       float* __restrict__ out,
                       _Float16* __restrict__ hbuf,
                       int* __restrict__ cnt)
{
    extern __shared__ __align__(16) char smem[];
    _Float16* sW = (_Float16*)smem;             // [64][KP]  weights (4 gates x 16 rows)
    _Float16* sC = sW + 64 * KP;                // [64][KP]  comb = [x_t, h_{t-1}]
    float*    sO = (float*)(sC + 64 * KP);      // [64][OP]  gate pre-activations
    float*    sB = sO + 64 * OP;                // [64]      biases

    const int tid   = threadIdx.x;
    const int wg    = blockIdx.x;
    const int hbase = wg * HSLICE;

    // ---- stage weight slice to LDS as f16 (once per launch) ----
    for (int it = 0; it < 64; ++it) {
        int e4  = it * THREADS + tid;           // float4 index, 256 per row
        int row = e4 >> 8;                      // 0..63 : gate*16 + r
        int k   = (e4 & 255) * 4;
        int gate = row >> 4, r = row & 15;
        const float* wp = (gate == 0) ? Wf : (gate == 1) ? Wi : (gate == 2) ? Wg : Wo;
        float4 v = *(const float4*)&wp[(size_t)(hbase + r) * KDIM + k];
        _Float16* d = &sW[row * KP + k];
        d[0] = (_Float16)v.x; d[1] = (_Float16)v.y;
        d[2] = (_Float16)v.z; d[3] = (_Float16)v.w;
    }
    if (tid < 64) {
        int gate = tid >> 4, j = tid & 15;
        const float* bp = (gate == 0) ? bf : (gate == 1) ? bi : (gate == 2) ? bg : bo;
        sB[tid] = bp[hbase + j];
    }

    // ---- wave / lane decomposition for WMMA tiles ----
    const int wv   = tid >> 5;                  // 0..7
    const int lane = tid & 31;
    const int mt   = wv & 3;                    // M tile (batch rows)
    const int nh   = wv >> 2;                   // 0/1 -> N tile pair
    const int n0   = nh * 2, n1 = n0 + 1;       // N tiles == gate index
    const int llo  = lane & 15, lhi = lane >> 4;

    // Fragment bases, this code's contiguous k = 16*half + s order (16B chunks per lane):
    // A 16x32 f16: lane = M row, low/high lane half split K by 16.
    const _Float16* aBase  = &sC[(mt * 16 + llo) * KP + lhi * 16];
    // B 32x16 f16: lane = N col, low/high lane half split K by 16.
    const _Float16* b0Base = &sW[(n0 * 16 + llo) * KP + lhi * 16];
    const _Float16* b1Base = &sW[(n1 * 16 + llo) * KP + lhi * 16];

    float creg[4] = {0.0f, 0.0f, 0.0f, 0.0f};   // cell state, register-resident
    int target = 0;

    for (int t = 0; t < T_STEPS; ++t) {
        const int rd = (t + 1) & 1;             // h double-buffer indices
        const int wr = t & 1;

        // ---- stage comb = [x_t | h_{t-1}] into LDS f16 ----
        const float* xt = x + (size_t)t * BATCH * IN_DIM;
        for (int it = 0; it < 32; ++it) {
            int e   = (it * THREADS + tid) * 4; // f32 element in x-part
            int row = e >> 9, k = e & 511;
            float4 v = *(const float4*)&xt[row * IN_DIM + k];
            _Float16* d = &sC[row * KP + k];
            d[0] = (_Float16)v.x; d[1] = (_Float16)v.y;
            d[2] = (_Float16)v.z; d[3] = (_Float16)v.w;
        }
        const _Float16* hsrc = hbuf + rd * BATCH * HID;
        for (int it = 0; it < 16; ++it) {
            int e   = (it * THREADS + tid) * 8; // f16 element in h-part
            int row = e >> 9, k = e & 511;
            *(v8h*)&sC[row * KP + IN_DIM + k] = *(const v8h*)&hsrc[row * HID + k];
        }
        __syncthreads();

        // ---- GEMM: [64 x 1024] x [1024 x 64] via v_wmma_f32_16x16x32_f16 ----
        v8f acc0 = {}; v8f acc1 = {};
#pragma unroll 4
        for (int kk = 0; kk < KDIM; kk += 32) {
            v8h a0 = *(const v8h*)(aBase + kk);
            v8h a1 = *(const v8h*)(aBase + kk + 8);
            v16h a = __builtin_shufflevector(a0, a1, 0,1,2,3,4,5,6,7,8,9,10,11,12,13,14,15);
            v8h p0 = *(const v8h*)(b0Base + kk);
            v8h p1 = *(const v8h*)(b0Base + kk + 8);
            v16h bm0 = __builtin_shufflevector(p0, p1, 0,1,2,3,4,5,6,7,8,9,10,11,12,13,14,15);
            v8h q0 = *(const v8h*)(b1Base + kk);
            v8h q1 = *(const v8h*)(b1Base + kk + 8);
            v16h bm1 = __builtin_shufflevector(q0, q1, 0,1,2,3,4,5,6,7,8,9,10,11,12,13,14,15);
            acc0 = __builtin_amdgcn_wmma_f32_16x16x32_f16(false, a, false, bm0, (short)0, acc0, false, false);
            acc1 = __builtin_amdgcn_wmma_f32_16x16x32_f16(false, a, false, bm1, (short)0, acc1, false, false);
        }

        // ---- spill D tiles to LDS (C/D layout: lane half adds M+8) ----
#pragma unroll
        for (int r = 0; r < 8; ++r) {
            int row = mt * 16 + lhi * 8 + r;
            sO[row * OP + n0 * 16 + llo] = acc0[r];
            sO[row * OP + n1 * 16 + llo] = acc1[r];
        }
        __syncthreads();

        // ---- elementwise gates + recurrence (c in registers) ----
        float*     outT = out + (size_t)t * BATCH * HID;
        _Float16*  hdst = hbuf + wr * BATCH * HID;
#pragma unroll
        for (int q = 0; q < 4; ++q) {
            int idx  = q * THREADS + tid;
            int brow = idx >> 4, j = idx & 15;
            const float* o_row = &sO[brow * OP];
            float fv = sig_ (o_row[      j] + sB[     j]);
            float iv = sig_ (o_row[16 +  j] + sB[16 + j]);
            float gv = tanh_(o_row[32 +  j] + sB[32 + j]);
            float ov = sig_ (o_row[48 +  j] + sB[48 + j]);
            float cv = fv * creg[q] + iv * gv;
            creg[q] = cv;
            float hv = ov * tanh_(cv);
            outT[brow * HID + hbase + j] = hv;
            hdst[brow * HID + hbase + j] = (_Float16)hv;
            if (t == T_STEPS - 1) {
                size_t tail = (size_t)T_STEPS * BATCH * HID;
                out[tail + brow * HID + hbase + j] = hv;                      // hx
                out[tail + (size_t)BATCH * HID + brow * HID + hbase + j] = cv; // cx
            }
        }

        // ---- grid-wide barrier (monotonic counter, one per step) ----
        target += NWG;
        __syncthreads();
        if (tid == 0) {
            __threadfence();
            __hip_atomic_fetch_add(cnt, 1, __ATOMIC_ACQ_REL, __HIP_MEMORY_SCOPE_AGENT);
            while (__hip_atomic_load(cnt, __ATOMIC_ACQUIRE, __HIP_MEMORY_SCOPE_AGENT) < target)
                __builtin_amdgcn_s_sleep(1);
            __threadfence();
        }
        __syncthreads();
    }
}

// ---------------------------------------------------------------------------
extern "C" void kernel_launch(void* const* d_in, const int* in_sizes, int n_in,
                              void* d_out, int out_size, void* d_ws, size_t ws_size,
                              hipStream_t stream) {
    (void)in_sizes; (void)n_in; (void)out_size; (void)ws_size;
    const float* x  = (const float*)d_in[0];
    const float* Wf = (const float*)d_in[1];
    const float* bf = (const float*)d_in[2];
    const float* Wi = (const float*)d_in[3];
    const float* bi = (const float*)d_in[4];
    const float* Wg = (const float*)d_in[5];
    const float* bg = (const float*)d_in[6];
    const float* Wo = (const float*)d_in[7];
    const float* bo = (const float*)d_in[8];
    float* out = (float*)d_out;

    _Float16* hbuf = (_Float16*)d_ws;                                   // 2*B*H f16
    int* cnt = (int*)((char*)d_ws + (size_t)2 * BATCH * HID * sizeof(_Float16));

    lstm_init_kernel<<<(2 * BATCH * HID + THREADS - 1) / THREADS, THREADS, 0, stream>>>(hbuf, cnt);

    const size_t smem_bytes = (size_t)64 * KP * sizeof(_Float16) * 2   // sW + sC
                            + (size_t)64 * OP * sizeof(float)          // sO
                            + 64 * sizeof(float);                      // sB
    lstm_persistent_kernel<<<NWG, THREADS, smem_bytes, stream>>>(
        x, Wf, bf, Wi, bi, Wg, bg, Wo, bo, out, hbuf, cnt);
}
